// LinearAttention_59803124629952
// MI455X (gfx1250) — matrix operand order repaired and checked by reading, not compiled, not run on hardware
//
#include <hip/hip_runtime.h>

typedef __attribute__((ext_vector_type(16))) _Float16 v16h;
typedef __attribute__((ext_vector_type(8)))  _Float16 v8h;
typedef __attribute__((ext_vector_type(8)))  float    v8f;

#define WMMA_F16(a, b, c) \
  __builtin_amdgcn_wmma_f32_16x16x32_f16(false, (a), false, (b), (short)0, (c), false, false)

static __device__ __forceinline__ v16h frag_cat(v8h lo, v8h hi) {
  v16h r;
#pragma unroll
  for (int i = 0; i < 8; ++i) { r[i] = lo[i]; r[i + 8] = hi[i]; }
  return r;
}

// A fragment: 16(M) x 32(K) f16, source row-major (stride in halves, multiple of 8, base 16B aligned)
// lane: m = lane&15; VGPR0-3 hold K = off..off+7, VGPR4-7 hold K = 16+off..16+off+7, off = (lane>=16)*8
static __device__ __forceinline__ v16h frag_a(const _Float16* base, int stride, int lane) {
  const _Float16* p = base + (lane & 15) * stride + ((lane >> 4) << 3);
  return frag_cat(*(const v8h*)p, *(const v8h*)(p + 16));
}

// B fragment: 32(K) x 16(N) f16 read from B^T (n-major rows, K contiguous)
// lane: n = lane&15; VGPR0-3 hold K = off..off+7, VGPR4-7 hold K = 16+off..16+off+7, off = (lane>=16)*8
static __device__ __forceinline__ v16h frag_b(const _Float16* baseT, int stride, int lane) {
  const _Float16* p = baseT + (lane & 15) * stride + ((lane >> 4) << 3);
  return frag_cat(*(const v8h*)p, *(const v8h*)(p + 16));
}

// ---------------------------------------------------------------------------
// Weight conversion: w_qkv (384x256) and w_out (256x128) f32 -> f16
__global__ void la_convert(const float* __restrict__ wqkv, const float* __restrict__ wout,
                           _Float16* __restrict__ wqkv_h, _Float16* __restrict__ wout_h) {
  int idx = blockIdx.x * 256 + threadIdx.x;
  if (idx < 98304) {
    wqkv_h[idx] = (_Float16)wqkv[idx];
  } else {
    int j = idx - 98304;
    if (j < 32768) wout_h[j] = (_Float16)wout[j];
  }
}

// ---------------------------------------------------------------------------
// Seed ksum / ctx with the mem_kv contributions (runs after memset, before stage1)
// mem_kv shape (2, 4 heads, 32, 4)
__global__ void la_memkv(const float* __restrict__ mem_kv,
                         float* __restrict__ ksum, float* __restrict__ ctx) {
  int b = blockIdx.x;
  int t = threadIdx.x;            // 128 threads: (h, d)
  int h = t >> 5, d = t & 31;
  const float* mk = mem_kv + ((size_t)h * 32 + d) * 4;
  const float* mv = mem_kv + 512 + (size_t)h * 32 * 4;
  float ek[4], s = 0.f;
#pragma unroll
  for (int j = 0; j < 4; ++j) { ek[j] = __expf(mk[j]); s += ek[j]; }
  ksum[b * 128 + t] += s;
  float* cb = ctx + (size_t)(b * 4 + h) * 1024 + d * 32;
  for (int e = 0; e < 32; ++e) {
    float cv = 0.f;
#pragma unroll
    for (int j = 0; j < 4; ++j) cv += ek[j] * mv[e * 4 + j];
    cb[e] += cv;
  }
}

// ---------------------------------------------------------------------------
// Stage 1: per (batch, 64-pixel tile)
//   RMS-norm(x) -> QKV GEMM (WMMA) -> q softmax(d)*scale -> qout (f16, [b][h][n][d])
//   expk row-sum partials -> atomicAdd ksum ; context partial = expk @ v^T (WMMA) -> atomicAdd ctx
__global__ __launch_bounds__(256) void la_stage1(
    const float* __restrict__ x, const float* __restrict__ g1,
    const _Float16* __restrict__ wqkv, _Float16* __restrict__ qout,
    float* __restrict__ ksum, float* __restrict__ ctx) {
  extern __shared__ __align__(16) char smem[];
  _Float16* xh  = (_Float16*)(smem);            // [64][264]  normalized x^T (f16)
  float*    qb  = (float*)(smem + 33792);       // [128][65]  q pre-softmax
  _Float16* kb  = (_Float16*)(smem + 67072);    // [128][72]  exp(k)
  _Float16* vb  = (_Float16*)(smem + 85504);    // [128][72]  v
  float*    part= (float*)(smem + 103936);      // [4][64]
  float*    inv = (float*)(smem + 104960);      // [64]
  float*    g1s = (float*)(smem + 105216);      // [256]

  const int t = threadIdx.x, lane = t & 31, wave = t >> 5;
  const int b = blockIdx.y;
  const int p0 = blockIdx.x * 64;

  g1s[t] = g1[t] * 16.0f;                       // fold sqrt(C)=16 into gamma
  __syncthreads();

  {  // load x tile (channel-strided), per-pixel sum of squares, store f16 x^T * g*16
    const int n = t & 63, cq = t >> 6;
    const float* xb = x + ((size_t)b * 256) * 4096 + p0 + n;
    float ss = 0.f;
#pragma unroll 4
    for (int i = 0; i < 64; ++i) {
      int c = i * 4 + cq;
      float v = xb[(size_t)c * 4096];
      ss += v * v;
      xh[n * 264 + c] = (_Float16)(v * g1s[c]);
    }
    part[cq * 64 + n] = ss;
  }
  __syncthreads();
  if (t < 64) {
    float s = part[t] + part[64 + t] + part[128 + t] + part[192 + t];
    inv[t] = 1.0f / fmaxf(sqrtf(s), 1e-12f);
  }
  __syncthreads();
  for (int n = 0; n < 64; ++n)
    xh[n * 264 + t] = (_Float16)((float)xh[n * 264 + t] * inv[n]);
  __syncthreads();

  // QKV GEMM: 24 row-tiles (M=384) x 4 col-tiles (N=64), K=256 in 8 steps, 12 tiles/wave
  for (int i = 0; i < 12; ++i) {
    int tid = wave + (i << 3);
    int mt = tid >> 2, nt = tid & 3;
    v8f acc = {};
    const _Float16* wb = wqkv + mt * 16 * 256;
    const _Float16* xb = xh + nt * 16 * 264;
#pragma unroll
    for (int ks = 0; ks < 8; ++ks) {
      v16h a = frag_a(wb + ks * 32, 256, lane);
      v16h bf = frag_b(xb + ks * 32, 264, lane);
      acc = WMMA_F16(a, bf, acc);
    }
    int mrb = mt * 16 + ((lane >> 4) << 3);
    int nc = nt * 16 + (lane & 15);
    if (mrb < 128) {                                   // q rows
#pragma unroll
      for (int r = 0; r < 8; ++r) qb[(mrb + r) * 65 + nc] = acc[r];
    } else if (mrb < 256) {                            // k rows -> exp(k), no max needed
#pragma unroll
      for (int r = 0; r < 8; ++r) {
        float ev = fminf(__expf(acc[r]), 60000.0f);
        kb[(mrb - 128 + r) * 72 + nc] = (_Float16)ev;
      }
    } else {                                           // v rows
#pragma unroll
      for (int r = 0; r < 8; ++r) vb[(mrb - 256 + r) * 72 + nc] = (_Float16)acc[r];
    }
  }
  __syncthreads();

  {  // q softmax over d=32 per (head, pixel); write qout[b][h][n][d] f16
    int hh = t >> 6, n = t & 63;
    float vals[32], s = 0.f;
#pragma unroll
    for (int d = 0; d < 32; ++d) {
      float e = __expf(qb[(hh * 32 + d) * 65 + n]);
      vals[d] = e; s += e;
    }
    float rs = 0.17677669529663687f / s;               // * dim_head^-0.5
    _Float16* qo = qout + (((size_t)(b * 4 + hh)) * 4096 + (p0 + n)) * 32;
#pragma unroll
    for (int d = 0; d < 32; ++d) qo[d] = (_Float16)(vals[d] * rs);
  }

  if (t < 128) {  // expk row-sum partial over this tile's 64 pixels
    float s = 0.f;
    for (int n = 0; n < 64; ++n) s += (float)kb[t * 72 + n];
    atomicAdd(&ksum[b * 128 + t], s);
  }

  // context partial: per head  expk(32x64) @ v^T(64x32)  -> atomicAdd into ctx[b][h][32][32]
#pragma unroll
  for (int i = 0; i < 2; ++i) {
    int tid = wave + (i << 3);                         // 16 tiles: (head, dt, et)
    int hh = tid >> 2, dt = (tid >> 1) & 1, et = tid & 1;
    v8f acc = {};
#pragma unroll
    for (int ks = 0; ks < 2; ++ks) {
      v16h a = frag_a(kb + (hh * 32 + dt * 16) * 72 + ks * 32, 72, lane);
      v16h bf = frag_b(vb + (hh * 32 + et * 16) * 72 + ks * 32, 72, lane);
      acc = WMMA_F16(a, bf, acc);
    }
    float* cb = ctx + (size_t)(b * 4 + hh) * 1024;
    int d0 = dt * 16 + ((lane >> 4) << 3);
    int e = et * 16 + (lane & 15);
#pragma unroll
    for (int r = 0; r < 8; ++r) atomicAdd(&cb[(d0 + r) * 32 + e], acc[r]);
  }
}

// ---------------------------------------------------------------------------
// Stage 3: per (batch, 64-pixel tile)
//   normalize ctx by ksum -> out = ctx^T @ q (WMMA per head) -> o = w_out @ out + b (WMMA)
//   -> RMS-norm over 256 channels -> final f32 output
__global__ __launch_bounds__(256) void la_stage3(
    const _Float16* __restrict__ qin, const float* __restrict__ ctx,
    const float* __restrict__ ksum, const _Float16* __restrict__ wout,
    const float* __restrict__ bout, const float* __restrict__ g2,
    float* __restrict__ out) {
  extern __shared__ __align__(16) char smem[];
  _Float16* ct  = (_Float16*)(smem);            // [4][32][40] ctx^T (e-major, d contiguous)
  _Float16* ot  = (_Float16*)(smem + 10240);    // [64][136]   out^T (n-major, hidden contiguous)
  float*    of  = (float*)(smem + 27648);       // [64][257]   o^T
  float*    bia = (float*)(smem + 93440);       // [256]
  float*    g2s = (float*)(smem + 94464);       // [256]
  float*    inv2= (float*)(smem + 95488);       // [64]

  const int t = threadIdx.x, lane = t & 31, wave = t >> 5;
  const int b = blockIdx.y;
  const int p0 = blockIdx.x * 64;

  bia[t] = bout[t];
  g2s[t] = g2[t] * 16.0f;

#pragma unroll
  for (int i = 0; i < 16; ++i) {  // normalize + transpose ctx -> f16
    int idx = t + i * 256;
    int hh = idx >> 10, rem = idx & 1023;
    int d = rem >> 5, e = rem & 31;
    float v = ctx[(size_t)(b * 4 + hh) * 1024 + d * 32 + e] / ksum[b * 128 + hh * 32 + d];
    ct[(hh * 32 + e) * 40 + d] = (_Float16)v;
  }
  __syncthreads();

  // GEMM1: per head  ctx^T(32x32) @ q(32x64); q read straight from global as B fragments
#pragma unroll
  for (int i = 0; i < 4; ++i) {
    int tid = wave + (i << 3);                         // 32 tiles: (head, et, nt)
    int hh = tid >> 3, et = (tid >> 2) & 1, nt = tid & 3;
    v16h a = frag_a(ct + (hh * 32 + et * 16) * 40, 40, lane);
    v16h bf = frag_b(qin + ((size_t)(b * 4 + hh) * 4096 + p0 + nt * 16) * 32, 32, lane);
    v8f acc = {};
    acc = WMMA_F16(a, bf, acc);
    int m0 = hh * 32 + et * 16 + ((lane >> 4) << 3);
    int nl = nt * 16 + (lane & 15);
#pragma unroll
    for (int r = 0; r < 8; ++r) ot[nl * 136 + (m0 + r)] = (_Float16)acc[r];
  }
  __syncthreads();

  // GEMM2: o = w_out(256x128) @ out(128x64) + bias
#pragma unroll
  for (int i = 0; i < 8; ++i) {
    int tid = wave + (i << 3);                         // 64 tiles: 16 row x 4 col
    int mt = tid >> 2, nt = tid & 3;
    v8f acc = {};
#pragma unroll
    for (int ks = 0; ks < 4; ++ks) {
      v16h a = frag_a(wout + mt * 16 * 128 + ks * 32, 128, lane);
      v16h bf = frag_b(ot + (nt * 16) * 136 + ks * 32, 136, lane);
      acc = WMMA_F16(a, bf, acc);
    }
    int m0 = mt * 16 + ((lane >> 4) << 3);
    int nl = nt * 16 + (lane & 15);
#pragma unroll
    for (int r = 0; r < 8; ++r) of[nl * 257 + (m0 + r)] = acc[r] + bia[m0 + r];
  }
  __syncthreads();

  if (t < 64) {  // per-pixel RMS over 256 channels
    float s = 0.f;
    for (int c = 0; c < 256; ++c) { float v = of[t * 257 + c]; s += v * v; }
    inv2[t] = 1.0f / fmaxf(sqrtf(s), 1e-12f);
  }
  __syncthreads();

  {  // coalesced channel-strided store
    const int n = t & 63, cq = t >> 6;
    float* ob = out + ((size_t)b * 256) * 4096 + p0 + n;
    float iv = inv2[n];
#pragma unroll 4
    for (int i = 0; i < 64; ++i) {
      int c = i * 4 + cq;
      ob[(size_t)c * 4096] = of[n * 257 + c] * iv * g2s[c];
    }
  }
}

// ---------------------------------------------------------------------------
extern "C" void kernel_launch(void* const* d_in, const int* in_sizes, int n_in,
                              void* d_out, int out_size, void* d_ws, size_t ws_size,
                              hipStream_t stream) {
  const float* x    = (const float*)d_in[0];
  const float* g1   = (const float*)d_in[1];
  const float* mkv  = (const float*)d_in[2];
  const float* wqkv = (const float*)d_in[3];
  const float* wout = (const float*)d_in[4];
  const float* bout = (const float*)d_in[5];
  const float* g2   = (const float*)d_in[6];
  float* out = (float*)d_out;

  char* ws = (char*)d_ws;
  _Float16* wqkv_h = (_Float16*)(ws);                 // 196608 B
  _Float16* wout_h = (_Float16*)(ws + 196608);        //  65536 B
  _Float16* qbuf   = (_Float16*)(ws + 262144);        // 33554432 B (q softmaxed, f16)
  float*    ksum   = (float*)(ws + 33816576);         //  16384 B
  float*    ctx    = (float*)(ws + 33832960);         // 524288 B

  hipMemsetAsync(ksum, 0, 16384 + 524288, stream);
  la_convert<<<512, 256, 0, stream>>>(wqkv, wout, wqkv_h, wout_h);
  la_memkv<<<32, 128, 0, stream>>>(mkv, ksum, ctx);

  hipFuncSetAttribute((const void*)la_stage1, hipFuncAttributeMaxDynamicSharedMemorySize, 106240);
  hipFuncSetAttribute((const void*)la_stage3, hipFuncAttributeMaxDynamicSharedMemorySize, 95744);

  dim3 grid(64, 32);
  la_stage1<<<grid, 256, 106240, stream>>>(x, g1, wqkv_h, qbuf, ksum, ctx);
  la_stage3<<<grid, 256, 95744, stream>>>(qbuf, ctx, ksum, wout_h, bout, g2, out);

  (void)in_sizes; (void)n_in; (void)out_size; (void)ws_size;
}
